// AlignmentModule_44306882625803
// MI455X (gfx1250) — matrix operand order repaired and checked by reading, not compiled, not run on hardware
//
#include <hip/hip_runtime.h>
#include <hip/hip_bf16.h>
#include <stdint.h>

// AlignmentModule: offset-conv (fp32 VALU) + deformable sampling (fp32 VALU)
// + main einsum as f16 WMMA (f32 accumulate) on gfx1250.
//
// Shapes: B=4, C=64, H=W=128, K=3 (9 taps), Cout=64.
// Main GEMM: D[M=65536, N=64] = A[M, K=576] * Bm[576, 64].

#define BATCH 4
#define CH    64
#define HH    128
#define WW    128
#define HW    (HH * WW)
#define TAPS  9
#define KDIM  (CH * TAPS)      // 576
#define NCO   64
#define NKC   (KDIM / 32)      // 18 K-chunks of 32
#define OFFC  (2 * TAPS)       // 18 offset channels
#define OFFSET_CLIP 10.0f

typedef __attribute__((ext_vector_type(16))) _Float16 v16h;
typedef __attribute__((ext_vector_type(8)))  _Float16 v8h;
typedef __attribute__((ext_vector_type(8)))  float    v8f;

// ---------------------------------------------------------------------------
// Kernel 1: offset = clip(conv3x3(concat(x, ref), offset_w) + offset_b, ±10)
// One thread per output pixel, 18 fp32 accumulators. Weight addressing is
// uniform across the block (OOB handled by zeroing v), enabling scalar loads.
// ---------------------------------------------------------------------------
__global__ void __launch_bounds__(256)
offset_conv_kernel(const float* __restrict__ x, const float* __restrict__ ref,
                   const float* __restrict__ ow, const float* __restrict__ ob,
                   float* __restrict__ offs) {
    const int idx = blockIdx.x * blockDim.x + threadIdx.x;   // 0 .. B*HW-1
    const int b = idx >> 14;
    const int h = (idx >> 7) & (HH - 1);
    const int w = idx & (WW - 1);

    float acc[OFFC];
#pragma unroll
    for (int co = 0; co < OFFC; ++co) acc[co] = ob[co];

    for (int ci = 0; ci < 2 * CH; ++ci) {
        const float* src = (ci < CH) ? (x   + ((size_t)b * CH + ci) * HW)
                                     : (ref + ((size_t)b * CH + (ci - CH)) * HW);
#pragma unroll
        for (int ky = 0; ky < 3; ++ky) {
#pragma unroll
            for (int kx = 0; kx < 3; ++kx) {
                const int iy = h - 1 + ky;
                const int ix = w - 1 + kx;
                const bool ok = (iy >= 0) & (iy < HH) & (ix >= 0) & (ix < WW);
                const int cy = min(max(iy, 0), HH - 1);
                const int cx = min(max(ix, 0), WW - 1);
                const float v = ok ? src[cy * WW + cx] : 0.0f;
                const float* wp = ow + ci * TAPS + ky * 3 + kx;   // uniform
#pragma unroll
                for (int co = 0; co < OFFC; ++co)
                    acc[co] += v * wp[co * (2 * CH * TAPS)];
            }
        }
    }
#pragma unroll
    for (int co = 0; co < OFFC; ++co) {
        float v = fminf(fmaxf(acc[co], -OFFSET_CLIP), OFFSET_CLIP);
        offs[((size_t)b * OFFC + co) * HW + h * WW + w] = v;
    }
}

// ---------------------------------------------------------------------------
// Kernel 2: pack weight[64][576] (fp32) into f16 WMMA B-fragments.
// Fragment f = kc*4 + nt (K-chunk kc in [0,18), Cout tile nt in [0,4)).
// Lane L: N = nt*16 + (L&15); lanes 0-15 hold K=kc*32+0..7,16..23, lanes 16-31
// hold K=kc*32+8..15,24..31; dword j packs a K-pair in that order. 256 dwords/frag.
// ---------------------------------------------------------------------------
__global__ void __launch_bounds__(256)
pack_weight_kernel(const float* __restrict__ wt, uint32_t* __restrict__ wsB) {
    const int t = blockIdx.x * blockDim.x + threadIdx.x;  // 0 .. 18*4*256-1
    const int f    = t >> 8;
    const int d    = t & 255;
    const int lane = d >> 3;
    const int j    = d & 7;
    const int kc = f >> 2;
    const int nt = f & 3;
    const int n  = nt * 16 + (lane & 15);
    const int kk = kc * 32 + ((lane & 16) ? 8 : 0) + ((j & 4) ? 16 : 0) + 2 * (j & 3);
    union { _Float16 h[2]; uint32_t u; } pk;
    pk.h[0] = (_Float16)wt[n * KDIM + kk];
    pk.h[1] = (_Float16)wt[n * KDIM + kk + 1];
    wsB[f * 256 + d] = pk.u;
}

// ---------------------------------------------------------------------------
// Kernel 3: deformable sampling (fp32 bilinear -> f16 A in LDS) + WMMA GEMM.
// Block = 128 threads (4 waves) per tile of 16 consecutive pixels (same row).
// All threads build A[16][576] f16 in LDS; each wave then computes one 16-wide
// Cout tile with 18 x v_wmma_f32_16x16x32_f16.
// ---------------------------------------------------------------------------
__global__ void __launch_bounds__(128)
deform_wmma_kernel(const float* __restrict__ x, const float* __restrict__ offs,
                   const uint32_t* __restrict__ wsB, float* __restrict__ out) {
    __shared__ __align__(16) _Float16 Asm[16 * KDIM];   // 18432 B

    const int tile = blockIdx.x;          // 0 .. B*HW/16 - 1
    const int P0 = tile * 16;
    const int b  = P0 >> 14;
    const int h  = (P0 >> 7) & (HH - 1);
    const int w0 = P0 & (WW - 1);         // 16 pixels stay within one row

    const int t = threadIdx.x;
    const int p = t & 15;                 // local pixel
    const int g = t >> 4;                 // channel group (8 channels each)
    const int w = w0 + p;

    const float* xb = x + (size_t)b * CH * HW;
    const float* ob = offs + (size_t)b * OFFC * HW + h * WW + w;

#pragma unroll
    for (int k = 0; k < TAPS; ++k) {
        const float dy = ob[(2 * k) * HW];
        const float dx = ob[(2 * k + 1) * HW];
        const float ys = (float)h - 1.0f + (float)(k / 3) + dy;
        const float xs = (float)w - 1.0f + (float)(k % 3) + dx;
        const float y0f = floorf(ys), x0f = floorf(xs);
        const float ty = ys - y0f, tx = xs - x0f;
        const int iy0 = (int)y0f, ix0 = (int)x0f;
        const int iy1 = iy0 + 1, ix1 = ix0 + 1;
        const float vy0 = (iy0 >= 0 && iy0 < HH) ? 1.0f : 0.0f;
        const float vy1 = (iy1 >= 0 && iy1 < HH) ? 1.0f : 0.0f;
        const float vx0 = (ix0 >= 0 && ix0 < WW) ? 1.0f : 0.0f;
        const float vx1 = (ix1 >= 0 && ix1 < WW) ? 1.0f : 0.0f;
        const int cy0 = min(max(iy0, 0), HH - 1), cx0 = min(max(ix0, 0), WW - 1);
        const int cy1 = min(max(iy1, 0), HH - 1), cx1 = min(max(ix1, 0), WW - 1);
        const float w00 = (1.0f - ty) * (1.0f - tx) * vy0 * vx0;
        const float w01 = (1.0f - ty) * tx * vy0 * vx1;
        const float w10 = ty * (1.0f - tx) * vy1 * vx0;
        const float w11 = ty * tx * vy1 * vx1;
        const int i00 = cy0 * WW + cx0, i01 = cy0 * WW + cx1;
        const int i10 = cy1 * WW + cx0, i11 = cy1 * WW + cx1;
#pragma unroll
        for (int cc = 0; cc < 8; ++cc) {
            const int c = g * 8 + cc;
            const float* xc = xb + (size_t)c * HW;
            const float v = w00 * xc[i00] + w01 * xc[i01] +
                            w10 * xc[i10] + w11 * xc[i11];
            Asm[p * KDIM + c * TAPS + k] = (_Float16)v;
        }
    }
    __syncthreads();

    // ---- WMMA GEMM: each wave owns Cout tile nt = wave id ----
    const int wave = t >> 5;
    const int lane = t & 31;
    v8f acc = {0.f, 0.f, 0.f, 0.f, 0.f, 0.f, 0.f, 0.f};

    const int arow = lane & 15;
    const int ahalf = (lane & 16) ? 8 : 0;   // K halves 8..15 / 24..31 side

    for (int kc = 0; kc < NKC; ++kc) {
        // A fragment from LDS (ISA 16-bit A layout):
        //   lanes 0-15 : K = 0..7 then 16..23 of this chunk
        //   lanes 16-31: K = 8..15 then 24..31
        const int abase = arow * KDIM + kc * 32 + ahalf;
        v8h a0 = *(const v8h*)&Asm[abase];          // ds_load_b128
        v8h a1 = *(const v8h*)&Asm[abase + 16];     // ds_load_b128
        v16h A = __builtin_shufflevector(a0, a1, 0, 1, 2, 3, 4, 5, 6, 7,
                                                 8, 9, 10, 11, 12, 13, 14, 15);
        // B fragment (pre-packed): 2 x 16B per lane
        const uint4* bp = reinterpret_cast<const uint4*>(
            wsB + ((kc * 4 + wave) << 8) + (lane << 3));
        union { uint4 q[2]; v16h h; } bu;
        bu.q[0] = bp[0];
        bu.q[1] = bp[1];

        acc = __builtin_amdgcn_wmma_f32_16x16x32_f16(
            /*neg_a=*/false, A, /*neg_b=*/false, bu.h,
            /*c_mod=*/(short)0, acc, /*reuse_a=*/false, /*reuse_b=*/false);
    }

    // D layout: VGPR r -> M = r + 8*(lane>=16), N = lane & 15
    const int n  = lane & 15;
    const int co = wave * 16 + n;
    const int mAdd = (lane & 16) ? 8 : 0;
    float* op = out + ((size_t)b * NCO + co) * HW + h * WW + w0;
#pragma unroll
    for (int r = 0; r < 8; ++r) op[mAdd + r] = acc[r];
}

// ---------------------------------------------------------------------------
extern "C" void kernel_launch(void* const* d_in, const int* in_sizes, int n_in,
                              void* d_out, int out_size, void* d_ws, size_t ws_size,
                              hipStream_t stream) {
    const float* x   = (const float*)d_in[0];
    const float* ref = (const float*)d_in[1];
    const float* ow  = (const float*)d_in[2];
    const float* obi = (const float*)d_in[3];
    const float* wt  = (const float*)d_in[4];
    float* out = (float*)d_out;

    // workspace: [0, 4.5MB) fp32 offsets, then 72KB of f16 B-fragments
    const size_t offsBytes = (size_t)BATCH * OFFC * HW * sizeof(float);
    float*    offs = (float*)d_ws;
    uint32_t* wsB  = (uint32_t*)((char*)d_ws + offsBytes);

    // K1: offset conv  (B*HW = 65536 pixels)
    offset_conv_kernel<<<(BATCH * HW) / 256, 256, 0, stream>>>(x, ref, ow, obi, offs);
    // K2: pack weights into WMMA B fragments (18*4 frags * 256 dwords)
    pack_weight_kernel<<<(NKC * 4 * 256) / 256, 256, 0, stream>>>(wt, wsB);
    // K3: deformable sampling + WMMA einsum (65536/16 = 4096 tiles)
    deform_wmma_kernel<<<(BATCH * HW) / 16, 128, 0, stream>>>(x, offs, wsB, out);
}
